// FlashAttention_34368328302999
// MI455X (gfx1250) — matrix operand order repaired and checked by reading, not compiled, not run on hardware
//
#include <hip/hip_runtime.h>

typedef __attribute__((ext_vector_type(16))) _Float16 v16h;
typedef __attribute__((ext_vector_type(8)))  float    v8f;

#define Bn   4
#define Ln   4096
#define Dn   1024
#define Hn   16
#define HDn  64
#define WINn 256
#define NCn  16
#define Mrows (Bn*Ln)        // 16384
#define N3D  (3*Dn)          // 3072
#define NITER 50

// ---------------- WMMA fragment helpers (contiguous k order, same for A and B) ----

// A-matrix 16x32 f16: lane L -> M = mBase + L%16; elem j: K = kBase + (L/16)*16 + j
__device__ __forceinline__ v16h load_a_frag(const _Float16* base, int ld,
                                            int mBase, int kBase, int lane) {
  const _Float16* p = base + (size_t)(mBase + (lane & 15)) * ld + kBase + ((lane >> 4) << 4);
  v16h a;
#pragma unroll
  for (int j = 0; j < 16; ++j) a[j] = p[j];
  return a;
}

// B-matrix 32x16 f16 from row-major [n][k] storage: lane L -> N = nBase + L%16;
// elem j: K = kBase + (L/16)*16 + j  (16 contiguous f16 = 32B per lane)
__device__ __forceinline__ v16h load_b_frag(const _Float16* base, int ld,
                                            int nBase, int kBase, int lane) {
  const _Float16* p = base + (size_t)(nBase + (lane & 15)) * ld + kBase + ((lane >> 4) << 4);
  v16h b;
#pragma unroll
  for (int j = 0; j < 16; ++j) b[j] = p[j];
  return b;
}

__device__ __forceinline__ v8f wmma32(v16h a, v16h b, v8f c) {
  return __builtin_amdgcn_wmma_f32_16x16x32_f16(false, a, false, b, (short)0, c,
                                                false, false);
}

// clamp(x,0)^e ; SQ path: e == 2 exactly (alpha = 1.5) -> v_max + v_mul/fma only
template <bool SQ>
__device__ __forceinline__ float pf(float x, float e) {
  x = fmaxf(x, 0.0f);
  if (SQ) return x * x;
  return (x > 0.0f) ? exp2f(e * log2f(x)) : 0.0f;
}

__device__ __forceinline__ void st4h(_Float16* dst, float4 v) {
  dst[0] = (_Float16)v.x; dst[1] = (_Float16)v.y;
  dst[2] = (_Float16)v.z; dst[3] = (_Float16)v.w;
}

// ---------------- entmax-bisect over 8 rows/half-wave, 16 col-tiles ----------
template <bool SQ>
__device__ __forceinline__ void entmax_rows(v8f (&xs)[16], float inv_am1, float lo_off) {
  float tau_lo[8], dm[8], f_lo[8], tau_m[8];
#pragma unroll
  for (int r = 0; r < 8; ++r) {
    float mx = -3.4e38f;
#pragma unroll
    for (int t = 0; t < 16; ++t) mx = fmaxf(mx, xs[t][r]);
    mx = fmaxf(mx, __shfl_xor(mx, 1));
    mx = fmaxf(mx, __shfl_xor(mx, 2));
    mx = fmaxf(mx, __shfl_xor(mx, 4));
    mx = fmaxf(mx, __shfl_xor(mx, 8));
    tau_lo[r] = mx - 1.0f;
    float s = 0.0f;
#pragma unroll
    for (int t = 0; t < 16; ++t) s += pf<SQ>(xs[t][r] - tau_lo[r], inv_am1);
    s += __shfl_xor(s, 1); s += __shfl_xor(s, 2);
    s += __shfl_xor(s, 4); s += __shfl_xor(s, 8);
    f_lo[r] = s - 1.0f;
    dm[r] = (mx - lo_off) - tau_lo[r];
    tau_m[r] = tau_lo[r];
  }
  for (int it = 0; it < NITER; ++it) {
#pragma unroll
    for (int r = 0; r < 8; ++r) {
      dm[r] *= 0.5f;
      float tm = tau_lo[r] + dm[r];
      float s = 0.0f;
#pragma unroll
      for (int t = 0; t < 16; ++t) s += pf<SQ>(xs[t][r] - tm, inv_am1);
      s += __shfl_xor(s, 1); s += __shfl_xor(s, 2);
      s += __shfl_xor(s, 4); s += __shfl_xor(s, 8);
      s -= 1.0f;
      if (s * f_lo[r] >= 0.0f) tau_lo[r] = tm;
      tau_m[r] = tm;
    }
  }
#pragma unroll
  for (int r = 0; r < 8; ++r) {
    float s = 0.0f;
#pragma unroll
    for (int t = 0; t < 16; ++t) {
      float p = pf<SQ>(xs[t][r] - tau_m[r], inv_am1);
      xs[t][r] = p;
      s += p;
    }
    s += __shfl_xor(s, 1); s += __shfl_xor(s, 2);
    s += __shfl_xor(s, 4); s += __shfl_xor(s, 8);
    float rp = 1.0f / s;
#pragma unroll
    for (int t = 0; t < 16; ++t) xs[t][r] *= rp;
  }
}

// ---------------- Kernel 1: qkv = x @ qkv_w.T + b  -> f16 workspace ----------
__global__ __launch_bounds__(256) void qkv_gemm(const float* __restrict__ x,
                                                const float* __restrict__ w,
                                                const float* __restrict__ bias,
                                                _Float16* __restrict__ out) {
  __shared__ _Float16 As[128][72];   // 128 x 64 (+8 pad), K-step 64
  __shared__ _Float16 Bs[128][72];
  const int m0 = blockIdx.x * 128;
  const int n0 = blockIdx.y * 128;
  const int tid = threadIdx.x;
  const int lane = tid & 31;
  const int wv = tid >> 5;       // 8 waves
  const int wm = wv & 3;         // 4 M-strips of 32 rows
  const int wn = wv >> 2;        // 2 N-strips of 64 cols

  v8f acc[2][4] = {};
  for (int k0 = 0; k0 < Dn; k0 += 64) {
#pragma unroll
    for (int i = 0; i < 8; ++i) {
      int idx = tid + i * 256;              // 2048 float4 chunks per matrix
      int r = idx >> 4, c4 = (idx & 15) << 2;
      float4 vx = *(const float4*)&x[(size_t)(m0 + r) * Dn + k0 + c4];
      float4 vw = *(const float4*)&w[(size_t)(n0 + r) * Dn + k0 + c4];
      st4h(&As[r][c4], vx);
      st4h(&Bs[r][c4], vw);
      if ((k0 + 64 < Dn) && ((i & 1) == 0)) {
        __builtin_prefetch(&x[(size_t)(m0 + r) * Dn + k0 + 64 + c4], 0, 1);
        __builtin_prefetch(&w[(size_t)(n0 + r) * Dn + k0 + 64 + c4], 0, 1);
      }
    }
    __syncthreads();
#pragma unroll
    for (int ks = 0; ks < 64; ks += 32) {
      v16h a0 = load_a_frag(&As[0][0], 72, wm * 32, ks, lane);
      v16h a1 = load_a_frag(&As[0][0], 72, wm * 32 + 16, ks, lane);
#pragma unroll
      for (int nt = 0; nt < 4; ++nt) {
        v16h b = load_b_frag(&Bs[0][0], 72, wn * 64 + nt * 16, ks, lane);
        acc[0][nt] = wmma32(a0, b, acc[0][nt]);
        acc[1][nt] = wmma32(a1, b, acc[1][nt]);
      }
    }
    __syncthreads();
  }
  const int l16 = lane & 15, hf = lane >> 4;
#pragma unroll
  for (int mt = 0; mt < 2; ++mt)
#pragma unroll
    for (int nt = 0; nt < 4; ++nt) {
      int n = n0 + wn * 64 + nt * 16 + l16;
      float bn = bias[n];
#pragma unroll
      for (int r = 0; r < 8; ++r) {
        int m = m0 + wm * 32 + mt * 16 + r + hf * 8;
        out[(size_t)m * N3D + n] = (_Float16)(acc[mt][nt][r] + bn);
      }
    }
}

// ---------------- Kernel 2: windowed entmax attention -----------------------
// one block per (b,h,window); 512 threads = 16 waves; wave w owns 16 query rows
__global__ __launch_bounds__(512) void attn_kernel(const _Float16* __restrict__ qkv,
                                                   const float* __restrict__ alpha,
                                                   _Float16* __restrict__ attn_out) {
  extern __shared__ char smem[];
  _Float16* Qs   = (_Float16*)smem;          // [256][72]
  _Float16* Ks   = Qs + 256 * 72;            // [256][72]
  _Float16* Vt   = Ks + 256 * 72;            // [64][264]  (transposed V)
  _Float16* Wscr = Vt + 64 * 264;            // [16 waves][16][40]

  const int blk = blockIdx.x;
  const int c = blk & (NCn - 1);
  const int h = (blk >> 4) & (Hn - 1);
  const int b = blk >> 8;
  const int tid = threadIdx.x;
  const int lane = tid & 31;
  const int wv = tid >> 5;
  const int l16 = lane & 15, hf = lane >> 4;

  const size_t rowBase = ((size_t)(b * Ln + c * WINn)) * N3D + h * (3 * HDn);
  // Q/K: rows are 128B contiguous f16 -> uint4 (8-half) copies
#pragma unroll
  for (int i = 0; i < 4; ++i) {
    int idx = tid + i * 512;                 // 2048 chunks = 256 rows * 8 chunks
    int r = idx >> 3, c8 = (idx & 7) << 3;
    const _Float16* src = qkv + rowBase + (size_t)r * N3D;
    *(uint4*)&Qs[r * 72 + c8] = *(const uint4*)&src[c8];
    *(uint4*)&Ks[r * 72 + c8] = *(const uint4*)&src[64 + c8];
  }
  // V transposed into LDS (scalar stores)
#pragma unroll
  for (int i = 0; i < 32; ++i) {
    int idx = tid + i * 512;                 // 16384 elements
    int r = idx >> 6, d = idx & 63;
    Vt[d * 264 + r] = qkv[rowBase + (size_t)r * N3D + 128 + d];
  }
  __syncthreads();

  const float am1 = alpha[h] - 1.0f;
  const float inv_am1 = 1.0f / am1;
  const float scale = am1 * 0.125f;          // am1 / sqrt(HD)
  const float lo_off = exp2f(-8.0f * am1);   // (1/256)^(alpha-1)

  // scores tile: 16 rows x 256 keys in 16 WMMA accumulators
  v16h aq0 = load_a_frag(Qs, 72, wv * 16, 0, lane);
  v16h aq1 = load_a_frag(Qs, 72, wv * 16, 32, lane);
  v8f xs[16];
#pragma unroll
  for (int t = 0; t < 16; ++t) {
    v16h b0 = load_b_frag(Ks, 72, t * 16, 0, lane);
    v8f az = {};
    az = wmma32(aq0, b0, az);
    v16h b1 = load_b_frag(Ks, 72, t * 16, 32, lane);
    xs[t] = wmma32(aq1, b1, az);
  }
#pragma unroll
  for (int t = 0; t < 16; ++t)
#pragma unroll
    for (int r = 0; r < 8; ++r) xs[t][r] *= scale;  // Xs = scores * (alpha-1)

  // entmax bisection; alpha = 1.5 -> exponent 2 -> pure mul/fma fast path
  if (fabsf(inv_am1 - 2.0f) < 1e-5f)
    entmax_rows<true>(xs, inv_am1, lo_off);
  else
    entmax_rows<false>(xs, inv_am1, lo_off);

  // ---- out = W @ V : re-layout C-format weights -> A-format via LDS scratch
  _Float16* myW = Wscr + wv * 16 * 40;
  v8f oacc[4] = {};
  for (int s = 0; s < 8; ++s) {              // 8 key-steps of 32
#pragma unroll
    for (int tt = 0; tt < 2; ++tt) {
      int t = 2 * s + tt;
#pragma unroll
      for (int r = 0; r < 8; ++r) {
        int m = r + hf * 8;
        myW[m * 40 + tt * 16 + l16] = (_Float16)xs[t][r];
      }
    }
    asm volatile("s_wait_dscnt 0" ::: "memory");   // per-wave LDS RAW fence
    v16h aw = load_a_frag(myW, 40, 0, 0, lane);
#pragma unroll
    for (int ft = 0; ft < 4; ++ft) {
      v16h bv = load_b_frag(Vt, 264, ft * 16, s * 32, lane);
      oacc[ft] = wmma32(aw, bv, oacc[ft]);
    }
  }
  const size_t obase = ((size_t)(b * Ln + c * WINn)) * Dn + h * HDn;
#pragma unroll
  for (int ft = 0; ft < 4; ++ft)
#pragma unroll
    for (int r = 0; r < 8; ++r) {
      int m = wv * 16 + r + hf * 8;
      attn_out[obase + (size_t)m * Dn + ft * 16 + l16] = (_Float16)oacc[ft][r];
    }
}

// ---------------- Kernel 3: per-row LayerNorm stats --------------------------
__global__ __launch_bounds__(256) void ln_stats(const _Float16* __restrict__ attn,
                                                float* __restrict__ stats) {
  const int row = blockIdx.x;
  const _Float16* p = attn + (size_t)row * Dn;
  const int tid = threadIdx.x, lane = tid & 31, wv = tid >> 5;
  float s = 0.0f, ss = 0.0f;
  for (int i = tid; i < Dn; i += 256) {
    float v = (float)p[i];
    s += v; ss += v * v;
  }
  s += __shfl_xor(s, 16); ss += __shfl_xor(ss, 16);
  s += __shfl_xor(s, 8);  ss += __shfl_xor(ss, 8);
  s += __shfl_xor(s, 4);  ss += __shfl_xor(ss, 4);
  s += __shfl_xor(s, 2);  ss += __shfl_xor(ss, 2);
  s += __shfl_xor(s, 1);  ss += __shfl_xor(ss, 1);
  __shared__ float red[16];
  if (lane == 0) { red[wv] = s; red[8 + wv] = ss; }
  __syncthreads();
  if (tid == 0) {
    float ts = 0.0f, tss = 0.0f;
    for (int i = 0; i < 8; ++i) { ts += red[i]; tss += red[8 + i]; }
    float mu = ts * (1.0f / Dn);
    float var = tss * (1.0f / Dn) - mu * mu;
    stats[row * 2]     = mu;
    stats[row * 2 + 1] = rsqrtf(var + 1e-5f);
  }
}

// ---------------- Kernel 4: out = LN(attn) @ o_w.T + o_b  (LN fused in stage)
__global__ __launch_bounds__(256) void out_gemm(const _Float16* __restrict__ attn,
                                                const float* __restrict__ stats,
                                                const float* __restrict__ gamma,
                                                const float* __restrict__ beta,
                                                const float* __restrict__ ow,
                                                const float* __restrict__ ob,
                                                float* __restrict__ out) {
  __shared__ _Float16 As[128][72];
  __shared__ _Float16 Bs[128][72];
  const int m0 = blockIdx.x * 128;
  const int n0 = blockIdx.y * 128;
  const int tid = threadIdx.x;
  const int lane = tid & 31;
  const int wv = tid >> 5;
  const int wm = wv & 3;
  const int wn = wv >> 2;

  v8f acc[2][4] = {};
  for (int k0 = 0; k0 < Dn; k0 += 64) {
    // A: LN applied on the fly, 8 halves per thread-chunk
#pragma unroll
    for (int i = 0; i < 4; ++i) {
      int idx = tid + i * 256;              // 1024 chunks = 128 rows * 8
      int r = idx >> 3, c8 = (idx & 7) << 3;
      int row = m0 + r, k = k0 + c8;
      float mu = stats[row * 2], rs = stats[row * 2 + 1];
      const _Float16* ap = &attn[(size_t)row * Dn + k];
#pragma unroll
      for (int j = 0; j < 8; ++j) {
        float v = (float)ap[j];
        As[r][c8 + j] = (_Float16)((v - mu) * rs * gamma[k + j] + beta[k + j]);
      }
    }
    // B: o_w row-major f32 -> f16, float4 chunks
#pragma unroll
    for (int i = 0; i < 8; ++i) {
      int idx = tid + i * 256;
      int r = idx >> 4, c4 = (idx & 15) << 2;
      float4 vw = *(const float4*)&ow[(size_t)(n0 + r) * Dn + k0 + c4];
      st4h(&Bs[r][c4], vw);
      if ((k0 + 64 < Dn) && ((i & 1) == 0))
        __builtin_prefetch(&ow[(size_t)(n0 + r) * Dn + k0 + 64 + c4], 0, 1);
    }
    __syncthreads();
#pragma unroll
    for (int ks = 0; ks < 64; ks += 32) {
      v16h a0 = load_a_frag(&As[0][0], 72, wm * 32, ks, lane);
      v16h a1 = load_a_frag(&As[0][0], 72, wm * 32 + 16, ks, lane);
#pragma unroll
      for (int nt = 0; nt < 4; ++nt) {
        v16h b = load_b_frag(&Bs[0][0], 72, wn * 64 + nt * 16, ks, lane);
        acc[0][nt] = wmma32(a0, b, acc[0][nt]);
        acc[1][nt] = wmma32(a1, b, acc[1][nt]);
      }
    }
    __syncthreads();
  }
  const int l16 = lane & 15, hf = lane >> 4;
#pragma unroll
  for (int mt = 0; mt < 2; ++mt)
#pragma unroll
    for (int nt = 0; nt < 4; ++nt) {
      int n = n0 + wn * 64 + nt * 16 + l16;
      float bn = ob[n];
#pragma unroll
      for (int r = 0; r < 8; ++r) {
        int m = m0 + wm * 32 + mt * 16 + r + hf * 8;
        out[(size_t)m * Dn + n] = acc[mt][nt][r] + bn;
      }
    }
}

// ---------------- launch -----------------------------------------------------
extern "C" void kernel_launch(void* const* d_in, const int* in_sizes, int n_in,
                              void* d_out, int out_size, void* d_ws, size_t ws_size,
                              hipStream_t stream) {
  const float* x     = (const float*)d_in[0];
  const float* alpha = (const float*)d_in[1];
  const float* qkv_w = (const float*)d_in[2];
  const float* qkv_b = (const float*)d_in[3];
  const float* o_w   = (const float*)d_in[4];
  const float* o_b   = (const float*)d_in[5];
  const float* ln_g  = (const float*)d_in[6];
  const float* ln_b  = (const float*)d_in[7];
  float* out = (float*)d_out;

  char* ws = (char*)d_ws;
  _Float16* qkv_h  = (_Float16*)ws;                                 // 100663296 B
  _Float16* attn_h = (_Float16*)(ws + (size_t)100663296);           //  33554432 B
  float*    stats  = (float*)(ws + (size_t)134217728);              //    131072 B

  qkv_gemm<<<dim3(Mrows / 128, N3D / 128), 256, 0, stream>>>(x, qkv_w, qkv_b, qkv_h);

  const size_t attn_smem = (size_t)(256 * 72 + 256 * 72 + 64 * 264 + 16 * 16 * 40) * 2;
  attn_kernel<<<Bn * Hn * NCn, 512, attn_smem, stream>>>(qkv_h, alpha, attn_h);

  ln_stats<<<Mrows, 256, 0, stream>>>(attn_h, stats);

  out_gemm<<<dim3(Mrows / 128, Dn / 128), 256, 0, stream>>>(attn_h, stats, ln_g, ln_b,
                                                            o_w, o_b, out);
}
